// HierarchicalModule_5523327943171
// MI455X (gfx1250) — compile-verified, matrix operand order checked by reading, not run on hardware
//
#include <hip/hip_runtime.h>

// ---------------------------------------------------------------------------
// MI455X (gfx1250) implementation of the hierarchical complex-attention module.
// All matmuls run on v_wmma_f32_16x16x32_bf16 (bf16 operands, fp32 accum).
// GEMM: 8 waves/block, 32x64 wave tile, W tile double-buffered through LDS
// via global_load_async_to_lds_b128 (ASYNCcnt path); K-loop unrolled 2x with
// register ping-pong so no VALU touches WMMA operands (no hazard NOPs).
// Attention: flash-style online softmax, P transposed through LDS, V
// pre-transposed so every WMMA operand load is a contiguous 32B v16bf.
// ---------------------------------------------------------------------------

typedef __bf16        v16bf __attribute__((ext_vector_type(16)));
typedef float         v8f   __attribute__((ext_vector_type(8)));
typedef unsigned int  v4u   __attribute__((ext_vector_type(4)));
typedef int           v4i   __attribute__((ext_vector_type(4)));

#define WMMA_BF16(a, b, c) \
  __builtin_amdgcn_wmma_f32_16x16x32_bf16(false, (a), false, (b), (short)0, (c), false, false)

#if defined(__has_builtin)
#if __has_builtin(__builtin_amdgcn_global_load_async_to_lds_b128)
#define HAVE_ASYNC_LDS 1
#endif
#endif
#ifndef HAVE_ASYNC_LDS
#define HAVE_ASYNC_LDS 0
#endif

#define GAS1 __attribute__((address_space(1)))
#define LAS3 __attribute__((address_space(3)))

static __device__ __forceinline__ void wait_async_lds() {
#if HAVE_ASYNC_LDS
#if __has_builtin(__builtin_amdgcn_s_wait_asynccnt)
  __builtin_amdgcn_s_wait_asynccnt(0);
#else
  asm volatile("s_wait_asynccnt 0x0" ::: "memory");
#endif
#endif
}

constexpr int   Bc  = 4;
constexpr int   Sc  = 1024;
constexpr int   Dc  = 1024;
constexpr int   Hc  = 16;
constexpr int   HDc = 64;
constexpr float SCALE = 0.125f;   // 1/sqrt(64)
constexpr float EPSc  = 1e-5f;

// ---------------------------------------------------------------------------
// fp32 -> bf16 elementwise convert (weights + pooled activations)
// ---------------------------------------------------------------------------
__global__ __launch_bounds__(256)
void f32_to_bf16_k(const float* __restrict__ in, __bf16* __restrict__ out) {
  const size_t i = (size_t)blockIdx.x * 256 + threadIdx.x;
  out[i] = (__bf16)in[i];
}

// ---------------------------------------------------------------------------
// LayerNorm over D=1024, bf16 output (feeds WMMA GEMMs)
// ---------------------------------------------------------------------------
__global__ __launch_bounds__(256)
void ln_bf16_k(const float* __restrict__ x, const float* __restrict__ g,
               const float* __restrict__ bta, __bf16* __restrict__ out) {
  __shared__ float sh[256];
  const int row = blockIdx.x;
  const int tid = threadIdx.x;
  const float* xr = x + (size_t)row * Dc;
  float vals[4];
  float s = 0.f, s2 = 0.f;
#pragma unroll
  for (int j = 0; j < 4; ++j) {
    vals[j] = xr[tid + 256 * j];
    s  += vals[j];
    s2 += vals[j] * vals[j];
  }
  sh[tid] = s; __syncthreads();
  for (int o = 128; o > 0; o >>= 1) { if (tid < o) sh[tid] += sh[tid + o]; __syncthreads(); }
  const float mean = sh[0] * (1.f / Dc);
  __syncthreads();
  sh[tid] = s2; __syncthreads();
  for (int o = 128; o > 0; o >>= 1) { if (tid < o) sh[tid] += sh[tid + o]; __syncthreads(); }
  const float var  = sh[0] * (1.f / Dc) - mean * mean;
  const float rstd = rsqrtf(var + EPSc);
#pragma unroll
  for (int j = 0; j < 4; ++j) {
    const int i = tid + 256 * j;
    out[(size_t)row * Dc + i] = (__bf16)((vals[j] - mean) * rstd * g[i] + bta[i]);
  }
}

// ---------------------------------------------------------------------------
// Fused (comb + upsampled-proj) add + LayerNorm, fp32 output to d_out
// ---------------------------------------------------------------------------
__global__ __launch_bounds__(256)
void final_add_ln_k(const float* __restrict__ comb, const float* __restrict__ up,
                    const float* __restrict__ g, const float* __restrict__ bta,
                    float* __restrict__ out) {
  __shared__ float sh[256];
  const int row = blockIdx.x;             // over B*S
  const int b = row / Sc, srow = row % Sc;
  const float* cr = comb + (size_t)row * Dc;
  const float* ur = up + ((size_t)(b * (Sc / 2) + (srow >> 1))) * Dc;
  const int tid = threadIdx.x;
  float vals[4];
  float s = 0.f, s2 = 0.f;
#pragma unroll
  for (int j = 0; j < 4; ++j) {
    const int i = tid + 256 * j;
    vals[j] = cr[i] + ur[i];
    s  += vals[j];
    s2 += vals[j] * vals[j];
  }
  sh[tid] = s; __syncthreads();
  for (int o = 128; o > 0; o >>= 1) { if (tid < o) sh[tid] += sh[tid + o]; __syncthreads(); }
  const float mean = sh[0] * (1.f / Dc);
  __syncthreads();
  sh[tid] = s2; __syncthreads();
  for (int o = 128; o > 0; o >>= 1) { if (tid < o) sh[tid] += sh[tid + o]; __syncthreads(); }
  const float var  = sh[0] * (1.f / Dc) - mean * mean;
  const float rstd = rsqrtf(var + EPSc);
#pragma unroll
  for (int j = 0; j < 4; ++j) {
    const int i = tid + 256 * j;
    out[(size_t)row * Dc + i] = (vals[j] - mean) * rstd * g[i] + bta[i];
  }
}

// ---------------------------------------------------------------------------
// bf16 WMMA GEMM:  C[M,N] = A[M,K] @ W[N,K]^T + bias
// 8 waves/block; wave tile 32x64; K-step 32, unrolled 2x (register ping-pong).
// W tile (64x32, 4KB) staged once per block per k-step into double-buffered
// LDS via async-to-LDS (fallback: register staging).
// Epilogue: outB!=null -> store bf16; else outF (+= if accum, = otherwise).
// Requires K % 64 == 0.
// ---------------------------------------------------------------------------
__global__ __launch_bounds__(256)
void gemm_bf16_k(const __bf16* __restrict__ A, const __bf16* __restrict__ W,
                 const float* __restrict__ bias,
                 __bf16* __restrict__ outB, float* __restrict__ outF,
                 int M, int N, int K, int accum) {
  __shared__ __align__(128) __bf16 wlds[2][64 * 32];
  const int tid  = threadIdx.x;
  const int lane = tid & 31;
  const int wv   = tid >> 5;
  const int m0   = (blockIdx.y * 8 + wv) * 32;
  const int n0   = blockIdx.x * 64;
  const int ar   = lane & 15;
  const int ah   = lane >> 4;

  // cooperative W-tile staging: thread -> one b128 (8 bf16) of the 64x32 tile
  const int wr = tid >> 2;          // 0..63 : output-channel row within tile
  const int wc = (tid & 3) * 8;     // 0,8,16,24 : k-offset within tile
  const __bf16* wsrc = W + (size_t)(n0 + wr) * K + wc;

  v8f c[2][4];
#pragma unroll
  for (int ms = 0; ms < 2; ++ms)
#pragma unroll
    for (int t = 0; t < 4; ++t)
#pragma unroll
      for (int v = 0; v < 8; ++v) c[ms][t][v] = 0.f;

  const __bf16* arow0 = A + (size_t)(m0 + ar) * K + ah * 16;
  const __bf16* arow1 = arow0 + (size_t)16 * K;

  // prologue: stage k=0 tile into buffer 0
#if HAVE_ASYNC_LDS
  __builtin_amdgcn_global_load_async_to_lds_b128(
      (GAS1 v4i*)(wsrc),
      (LAS3 v4i*)(&wlds[0][wr * 32 + wc]), 0, 0);
  wait_async_lds();
#else
  *(v4u*)(&wlds[0][wr * 32 + wc]) = *(const v4u*)wsrc;
#endif
  __syncthreads();

  v16bf a0 = *(const v16bf*)(arow0);
  v16bf a1 = *(const v16bf*)(arow1);

  for (int k0 = 0; k0 < K; k0 += 64) {
    // ---- substep A: compute on buffer 0 (tile k0); stage buffer 1 (k0+32)
#if HAVE_ASYNC_LDS
    __builtin_amdgcn_global_load_async_to_lds_b128(
        (GAS1 v4i*)(wsrc + k0 + 32),
        (LAS3 v4i*)(&wlds[1][wr * 32 + wc]), 0, 0);
#else
    const v4u wtmpA = *(const v4u*)(wsrc + k0 + 32);
#endif
    const v16bf b0 = *(const v16bf*)(arow0 + k0 + 32);
    const v16bf b1 = *(const v16bf*)(arow1 + k0 + 32);
#pragma unroll
    for (int t = 0; t < 4; ++t) {
      const v16bf wb = *(const v16bf*)(&wlds[0][(t * 16 + ar) * 32 + ah * 16]);
      c[0][t] = WMMA_BF16(a0, wb, c[0][t]);
      c[1][t] = WMMA_BF16(a1, wb, c[1][t]);
    }
#if HAVE_ASYNC_LDS
    wait_async_lds();
#else
    *(v4u*)(&wlds[1][wr * 32 + wc]) = wtmpA;
#endif
    __syncthreads();

    // ---- substep B: compute on buffer 1 (tile k0+32); stage buffer 0 (k0+64)
    const bool more = (k0 + 64) < K;
#if HAVE_ASYNC_LDS
    if (more)
      __builtin_amdgcn_global_load_async_to_lds_b128(
          (GAS1 v4i*)(wsrc + k0 + 64),
          (LAS3 v4i*)(&wlds[0][wr * 32 + wc]), 0, 0);
#else
    v4u wtmpB;
    if (more) wtmpB = *(const v4u*)(wsrc + k0 + 64);
#endif
    if (more) {
      a0 = *(const v16bf*)(arow0 + k0 + 64);
      a1 = *(const v16bf*)(arow1 + k0 + 64);
    }
#pragma unroll
    for (int t = 0; t < 4; ++t) {
      const v16bf wb = *(const v16bf*)(&wlds[1][(t * 16 + ar) * 32 + ah * 16]);
      c[0][t] = WMMA_BF16(b0, wb, c[0][t]);
      c[1][t] = WMMA_BF16(b1, wb, c[1][t]);
    }
#if HAVE_ASYNC_LDS
    wait_async_lds();
#else
    if (more) *(v4u*)(&wlds[0][wr * 32 + wc]) = wtmpB;
#endif
    __syncthreads();
  }

#pragma unroll
  for (int ms = 0; ms < 2; ++ms)
#pragma unroll
    for (int t = 0; t < 4; ++t)
#pragma unroll
      for (int v = 0; v < 8; ++v) {
        const int r = m0 + ms * 16 + v + 8 * ah;
        const int n = n0 + t * 16 + ar;
        float val = c[ms][t][v] + (bias ? bias[n] : 0.f);
        const size_t idx = (size_t)r * N + n;
        if (outB)        outB[idx] = (__bf16)val;
        else if (accum)  outF[idx] += val;
        else             outF[idx] = val;
      }
}

// ---------------------------------------------------------------------------
// V transpose: [B,S,D] (head-merged) -> VT [B*H, HD, S] so PV WMMA B-fragments
// are contiguous along the key dimension.
// ---------------------------------------------------------------------------
__global__ __launch_bounds__(256)
void vtrans_k(const __bf16* __restrict__ V, __bf16* __restrict__ VT, int Scur) {
  const size_t i = (size_t)blockIdx.x * 256 + threadIdx.x;   // over B*Scur*D
  const int d  = (int)(i % Dc);
  const size_t bs = i / Dc;
  const int srow = (int)(bs % Scur);
  const int b    = (int)(bs / Scur);
  const int h = d >> 6, ccol = d & 63;
  VT[(((size_t)(b * Hc + h) * HDc + ccol)) * Scur + srow] = V[i];
}

// ---------------------------------------------------------------------------
// Flash attention (complex scores): one wave per 16-query tile per (b,h).
//   scores = (Qr.Kr^T + Qi.Ki^T) * SCALE ; softmax online ; O = P@Vr, P@Vi
// 16 WMMAs per 32-key iteration; P transposed C-layout -> A-layout via LDS.
// ---------------------------------------------------------------------------
__global__ __launch_bounds__(32)
void attn_flash_k(const __bf16* __restrict__ QR, const __bf16* __restrict__ QI,
                  const __bf16* __restrict__ KR, const __bf16* __restrict__ KI,
                  const __bf16* __restrict__ VTR, const __bf16* __restrict__ VTI,
                  __bf16* __restrict__ ORE, __bf16* __restrict__ OIM, int Scur) {
  __shared__ __align__(64) __bf16 pbuf[16 * 32];
  const int l  = threadIdx.x;
  const int ar = l & 15;
  const int ah = l >> 4;
  const int q0 = blockIdx.x * 16;
  const int bh = blockIdx.y;
  const int b  = bh / Hc;
  const int h  = bh - b * Hc;
  const size_t qbase = ((size_t)(b * Scur + q0)) * Dc + h * HDc;

  // Q fragments: two 16x32 A-frags per matrix covering HD=64
  const v16bf qr0 = *(const v16bf*)(QR + qbase + (size_t)ar * Dc + ah * 16);
  const v16bf qr1 = *(const v16bf*)(QR + qbase + (size_t)ar * Dc + 32 + ah * 16);
  const v16bf qi0 = *(const v16bf*)(QI + qbase + (size_t)ar * Dc + ah * 16);
  const v16bf qi1 = *(const v16bf*)(QI + qbase + (size_t)ar * Dc + 32 + ah * 16);

  float mrow[8], lrow[8];
  v8f ore[4], oim[4];
#pragma unroll
  for (int v = 0; v < 8; ++v) { mrow[v] = -1e30f; lrow[v] = 0.f; }
#pragma unroll
  for (int t = 0; t < 4; ++t)
#pragma unroll
    for (int v = 0; v < 8; ++v) { ore[t][v] = 0.f; oim[t][v] = 0.f; }

  for (int kt = 0; kt < Scur; kt += 32) {
    v8f sc[2];
#pragma unroll
    for (int j = 0; j < 2; ++j) {
      const size_t kb = ((size_t)(b * Scur + kt + j * 16 + ar)) * Dc + h * HDc + ah * 16;
      const v16bf kr0 = *(const v16bf*)(KR + kb);
      const v16bf kr1 = *(const v16bf*)(KR + kb + 32);
      const v16bf ki0 = *(const v16bf*)(KI + kb);
      const v16bf ki1 = *(const v16bf*)(KI + kb + 32);
      v8f acc;
#pragma unroll
      for (int v = 0; v < 8; ++v) acc[v] = 0.f;
      acc = WMMA_BF16(qr0, kr0, acc);
      acc = WMMA_BF16(qr1, kr1, acc);
      acc = WMMA_BF16(qi0, ki0, acc);
      acc = WMMA_BF16(qi1, ki1, acc);
      sc[j] = acc * SCALE;
    }

    // Online softmax: row stats reduced across the 16-lane half (rows v, v+8)
#pragma unroll
    for (int v = 0; v < 8; ++v) {
      float tmax = fmaxf(sc[0][v], sc[1][v]);
#pragma unroll
      for (int o = 1; o < 16; o <<= 1) tmax = fmaxf(tmax, __shfl_xor(tmax, o, 32));
      const float mnew  = fmaxf(mrow[v], tmax);
      const float alpha = __expf(mrow[v] - mnew);
      mrow[v] = mnew;
      const float p0 = __expf(sc[0][v] - mnew);
      const float p1 = __expf(sc[1][v] - mnew);
      const int row = v + 8 * ah;
      pbuf[row * 32 + ar]      = (__bf16)p0;
      pbuf[row * 32 + 16 + ar] = (__bf16)p1;
      float rs = p0 + p1;
#pragma unroll
      for (int o = 1; o < 16; o <<= 1) rs += __shfl_xor(rs, o, 32);
      lrow[v] = lrow[v] * alpha + rs;
#pragma unroll
      for (int t = 0; t < 4; ++t) { ore[t][v] *= alpha; oim[t][v] *= alpha; }
    }
    __syncthreads();
    // Re-load P in the 16x32 A-fragment layout (LDS transpose)
    const v16bf pa = *(const v16bf*)(pbuf + ar * 32 + ah * 16);
#pragma unroll
    for (int t = 0; t < 4; ++t) {
      const size_t vtb = ((size_t)(bh * HDc + t * 16 + ar)) * Scur + kt + ah * 16;
      const v16bf vr = *(const v16bf*)(VTR + vtb);
      const v16bf vi = *(const v16bf*)(VTI + vtb);
      ore[t] = WMMA_BF16(pa, vr, ore[t]);
      oim[t] = WMMA_BF16(pa, vi, oim[t]);
    }
    __syncthreads();
  }

#pragma unroll
  for (int t = 0; t < 4; ++t)
#pragma unroll
    for (int v = 0; v < 8; ++v) {
      const int row = v + 8 * ah;
      const float inv = 1.0f / lrow[v];
      const size_t o = qbase + (size_t)row * Dc + t * 16 + ar;
      ORE[o] = (__bf16)(ore[t][v] * inv);
      OIM[o] = (__bf16)(oim[t][v] * inv);
    }
}

// ---------------------------------------------------------------------------
// pool2 / upsample-add (fp32)
// ---------------------------------------------------------------------------
__global__ __launch_bounds__(256)
void pool2_k(const float* __restrict__ in, float* __restrict__ out, int Shalf) {
  const size_t i = (size_t)blockIdx.x * 256 + threadIdx.x;   // over B*Shalf*D
  const int d = (int)(i % Dc);
  const size_t bs = i / Dc;
  const int s2 = (int)(bs % Shalf);
  const int b  = (int)(bs / Shalf);
  const size_t base = ((size_t)(b * 2 * Shalf + 2 * s2)) * Dc + d;
  out[i] = 0.5f * (in[base] + in[base + Dc]);
}

__global__ __launch_bounds__(256)
void upadd_k(float* __restrict__ acc, const float* __restrict__ up, int Scur) {
  const size_t i = (size_t)blockIdx.x * 256 + threadIdx.x;   // over B*Scur*D
  const int d = (int)(i % Dc);
  const size_t bs = i / Dc;
  const int srow = (int)(bs % Scur);
  const int b    = (int)(bs / Scur);
  acc[i] += up[((size_t)(b * (Scur / 2) + (srow >> 1))) * Dc + d];
}

// ---------------------------------------------------------------------------
// Host orchestration
// ---------------------------------------------------------------------------
extern "C" void kernel_launch(void* const* d_in, const int* in_sizes, int n_in,
                              void* d_out, int out_size, void* d_ws, size_t ws_size,
                              hipStream_t stream) {
  (void)in_sizes; (void)n_in; (void)out_size; (void)ws_size;

  const float* x_re   = (const float*)d_in[0];
  const float* x_im   = (const float*)d_in[1];
  const float* attn_W = (const float*)d_in[2];
  const float* attn_b = (const float*)d_in[3];
  const float* ln_g   = (const float*)d_in[4];
  const float* ln_b   = (const float*)d_in[5];
  const float* wp_W   = (const float*)d_in[6];
  const float* wp_b   = (const float*)d_in[7];
  const float* ps_W   = (const float*)d_in[8];
  const float* ps_b   = (const float*)d_in[9];
  float* outp = (float*)d_out;

  constexpr size_t MB = 1ull << 20;
  char* w = (char*)d_ws;
  // fp32 buffers
  float* A_re  = (float*)(w + 0 * MB);     // 16 MB  residual stream (word / comb)
  float* A_im  = (float*)(w + 16 * MB);    // 16 MB
  float* PH_re = (float*)(w + 32 * MB);    //  8 MB  phrase level
  float* PH_im = (float*)(w + 40 * MB);
  float* SE_re = (float*)(w + 48 * MB);    //  8 MB  sentence level
  float* SE_im = (float*)(w + 56 * MB);
  float* TMP_re = (float*)(w + 64 * MB);   //  8 MB  projection outputs
  float* TMP_im = (float*)(w + 72 * MB);
  // bf16 buffers
  __bf16* NRE = (__bf16*)(w + 80 * MB);    //  8 MB each
  __bf16* NIM = (__bf16*)(w + 88 * MB);
  __bf16* QR  = (__bf16*)(w + 96 * MB);
  __bf16* QI  = (__bf16*)(w + 104 * MB);
  __bf16* KR  = (__bf16*)(w + 112 * MB);
  __bf16* KI  = (__bf16*)(w + 120 * MB);
  __bf16* VR  = (__bf16*)(w + 128 * MB);
  __bf16* VI  = (__bf16*)(w + 136 * MB);
  __bf16* VTR = (__bf16*)(w + 144 * MB);
  __bf16* VTI = (__bf16*)(w + 152 * MB);
  __bf16* ORE = (__bf16*)(w + 160 * MB);
  __bf16* OIM = (__bf16*)(w + 168 * MB);
  __bf16* WBF = (__bf16*)(w + 176 * MB);   // 60 MB of bf16 weights (7*4 + 2 D*D)

  const size_t DD = (size_t)Dc * Dc;

  // 0) copy inputs into the residual stream; convert all weights to bf16
  (void)hipMemcpyAsync(A_re, x_re, (size_t)Bc * Sc * Dc * sizeof(float), hipMemcpyDeviceToDevice, stream);
  (void)hipMemcpyAsync(A_im, x_im, (size_t)Bc * Sc * Dc * sizeof(float), hipMemcpyDeviceToDevice, stream);
  f32_to_bf16_k<<<(unsigned)((28 * DD) / 256), 256, 0, stream>>>(attn_W, WBF);
  f32_to_bf16_k<<<(unsigned)(DD / 256), 256, 0, stream>>>(wp_W, WBF + 28 * DD);
  f32_to_bf16_k<<<(unsigned)(DD / 256), 256, 0, stream>>>(ps_W, WBF + 29 * DD);

  auto layer = [&](float* Are, float* Aim, int Scur, int li) {
    const __bf16* Wl = WBF + (size_t)li * 4 * DD;
    const float*  bl = attn_b + (size_t)li * 4 * Dc;
    const int Mrows = Bc * Scur;
    const unsigned nelem = (unsigned)((size_t)Mrows * Dc / 256);
    const dim3 gg(Dc / 64, Mrows / 256);

    ln_bf16_k<<<Mrows, 256, 0, stream>>>(Are, ln_g, ln_b, NRE);
    ln_bf16_k<<<Mrows, 256, 0, stream>>>(Aim, ln_g, ln_b, NIM);

    gemm_bf16_k<<<gg, 256, 0, stream>>>(NRE, Wl,          bl,          QR, nullptr, Mrows, Dc, Dc, 0);
    gemm_bf16_k<<<gg, 256, 0, stream>>>(NIM, Wl,          bl,          QI, nullptr, Mrows, Dc, Dc, 0);
    gemm_bf16_k<<<gg, 256, 0, stream>>>(NRE, Wl + DD,     bl + Dc,     KR, nullptr, Mrows, Dc, Dc, 0);
    gemm_bf16_k<<<gg, 256, 0, stream>>>(NIM, Wl + DD,     bl + Dc,     KI, nullptr, Mrows, Dc, Dc, 0);
    gemm_bf16_k<<<gg, 256, 0, stream>>>(NRE, Wl + 2 * DD, bl + 2 * Dc, VR, nullptr, Mrows, Dc, Dc, 0);
    gemm_bf16_k<<<gg, 256, 0, stream>>>(NIM, Wl + 2 * DD, bl + 2 * Dc, VI, nullptr, Mrows, Dc, Dc, 0);

    vtrans_k<<<nelem, 256, 0, stream>>>(VR, VTR, Scur);
    vtrans_k<<<nelem, 256, 0, stream>>>(VI, VTI, Scur);

    attn_flash_k<<<dim3(Scur / 16, Bc * Hc), 32, 0, stream>>>(QR, QI, KR, KI, VTR, VTI, ORE, OIM, Scur);

    gemm_bf16_k<<<gg, 256, 0, stream>>>(ORE, Wl + 3 * DD, bl + 3 * Dc, nullptr, Are, Mrows, Dc, Dc, 1);
    gemm_bf16_k<<<gg, 256, 0, stream>>>(OIM, Wl + 3 * DD, bl + 3 * Dc, nullptr, Aim, Mrows, Dc, Dc, 1);
  };

  // word level (S=1024)
  layer(A_re, A_im, Sc, 0);
  layer(A_re, A_im, Sc, 1);
  layer(A_re, A_im, Sc, 2);

  // phrase level (S=512)
  const int Sh = Sc / 2;
  const unsigned nh = (unsigned)((size_t)Bc * Sh * Dc / 256);
  const unsigned nf = (unsigned)((size_t)Bc * Sc * Dc / 256);
  pool2_k<<<nh, 256, 0, stream>>>(A_re, PH_re, Sh);
  pool2_k<<<nh, 256, 0, stream>>>(A_im, PH_im, Sh);
  layer(PH_re, PH_im, Sh, 3);
  layer(PH_re, PH_im, Sh, 4);
  layer(PH_re, PH_im, Sh, 5);

  // word<-phrase projection (bias on real part only) + upsample-add into comb
  {
    const dim3 gp(Dc / 64, (Bc * Sh) / 256);
    f32_to_bf16_k<<<nh, 256, 0, stream>>>(PH_re, NRE);
    f32_to_bf16_k<<<nh, 256, 0, stream>>>(PH_im, NIM);
    gemm_bf16_k<<<gp, 256, 0, stream>>>(NRE, WBF + 28 * DD, wp_b,    nullptr, TMP_re, Bc * Sh, Dc, Dc, 0);
    gemm_bf16_k<<<gp, 256, 0, stream>>>(NIM, WBF + 28 * DD, nullptr, nullptr, TMP_im, Bc * Sh, Dc, Dc, 0);
    upadd_k<<<nf, 256, 0, stream>>>(A_re, TMP_re, Sc);
    upadd_k<<<nf, 256, 0, stream>>>(A_im, TMP_im, Sc);
  }

  // sentence level (S=512, pooled from comb)
  pool2_k<<<nh, 256, 0, stream>>>(A_re, SE_re, Sh);
  pool2_k<<<nh, 256, 0, stream>>>(A_im, SE_im, Sh);
  layer(SE_re, SE_im, Sh, 6);

  // phrase<-sentence projection + final fused add + LayerNorm -> d_out
  {
    const dim3 gp(Dc / 64, (Bc * Sh) / 256);
    f32_to_bf16_k<<<nh, 256, 0, stream>>>(SE_re, NRE);
    f32_to_bf16_k<<<nh, 256, 0, stream>>>(SE_im, NIM);
    gemm_bf16_k<<<gp, 256, 0, stream>>>(NRE, WBF + 29 * DD, ps_b,    nullptr, TMP_re, Bc * Sh, Dc, Dc, 0);
    gemm_bf16_k<<<gp, 256, 0, stream>>>(NIM, WBF + 29 * DD, nullptr, nullptr, TMP_im, Bc * Sh, Dc, Dc, 0);
    final_add_ln_k<<<Bc * Sc, 256, 0, stream>>>(A_re, TMP_re, ln_g, ln_b, outp);
    final_add_ln_k<<<Bc * Sc, 256, 0, stream>>>(A_im, TMP_im, ln_g, ln_b, outp + (size_t)Bc * Sc * Dc);
  }
}
